// CNF_19181323944551
// MI455X (gfx1250) — matrix operand order repaired and checked by reading, not compiled, not run on hardware
//
#include <hip/hip_runtime.h>
#include <hip/hip_bf16.h>

typedef __attribute__((ext_vector_type(16))) _Float16 v16h;
typedef __attribute__((ext_vector_type(8)))  _Float16 v8h;
typedef __attribute__((ext_vector_type(8)))  float    v8f;

#define CNF_D 64
#define CNF_H 512
#define CNF_B 4096
#define CNF_NSTEPS 17
#define ROWS 16      // batch rows per block
#define BLK 128      // 4 wave32s

static __device__ __forceinline__ v16h cat16(v8h lo, v8h hi) {
  v16h r;
  #pragma unroll
  for (int i = 0; i < 8; ++i) { r[i] = lo[i]; r[8 + i] = hi[i]; }
  return r;
}

// Branch-free tanh: 1 - 2/(e^{2x}+1), saturates to +/-1, no EXEC divergence.
static __device__ __forceinline__ float fast_tanh(float v) {
  const float TWO_LOG2E = 2.8853900817779268f;  // 2*log2(e)
  float e = __builtin_amdgcn_exp2f(v * TWO_LOG2E);   // v_exp_f32
  float r = __builtin_amdgcn_rcpf(e + 1.f);          // v_rcp_f32
  return 1.f - 2.f * r;
}

// Swizzle fp32 row-major [K,N] into f16 WMMA B-operand tiles:
// tile (kt,nt) -> 32 lanes x 16 halves contiguous; lane holds
// (K = kt*32 + (lane/16)*8 + (h%8) + (h/8)*16, N = nt*16 + lane%16).
__global__ void cnf_prep_swizzle(const float* __restrict__ src, _Float16* __restrict__ dst,
                                 int K, int N) {
  int id = blockIdx.x * blockDim.x + threadIdx.x;
  if (id >= K * N) return;
  int NT = N >> 4;
  int tileId = id >> 9;
  int within = id & 511;
  int lane = within >> 4;
  int h = within & 15;
  int kt = tileId / NT;
  int nt = tileId % NT;
  int Kg = kt * 32 + (lane >> 4) * 8 + (h & 7) + (h >> 3) * 16;
  int Ng = nt * 16 + (lane & 15);
  dst[id] = (_Float16)src[Kg * N + Ng];
}

// C[j,k] = W2[j,k] * sum_d W1[d,j]*W3[k,d], emitted directly in swizzled B-layout.
__global__ void cnf_prep_cmat(const float* __restrict__ W1, const float* __restrict__ W2,
                              const float* __restrict__ W3, _Float16* __restrict__ dst) {
  int id = blockIdx.x * blockDim.x + threadIdx.x;
  if (id >= CNF_H * CNF_H) return;
  int tileId = id >> 9;
  int within = id & 511;
  int lane = within >> 4;
  int h = within & 15;
  int kt = tileId >> 5;   // NT = 32
  int nt = tileId & 31;
  int j = kt * 32 + (lane >> 4) * 8 + (h & 7) + (h >> 3) * 16;   // K index (layer-1 unit)
  int k = nt * 16 + (lane & 15);            // N index (layer-2 unit)
  float s = 0.f;
  #pragma unroll 8
  for (int d = 0; d < CNF_D; ++d) s += W1[d * CNF_H + j] * W3[k * CNF_D + d];
  dst[id] = (_Float16)(W2[j * CNF_H + k] * s);
}

__global__ __launch_bounds__(BLK)
void cnf_ode_kernel(const float* __restrict__ x, const float* __restrict__ tspan,
                    const float* __restrict__ W1, const float* __restrict__ b1,
                    const float* __restrict__ b2, const float* __restrict__ b3,
                    const _Float16* __restrict__ W1B, const _Float16* __restrict__ W2B,
                    const _Float16* __restrict__ W3B, const _Float16* __restrict__ CB,
                    float* __restrict__ outZ, float* __restrict__ outL) {
  __shared__ float    sS[ROWS][72];      // state  [x(64), logdet]
  __shared__ float    sXi[ROWS][72];     // stage input
  __shared__ float    sK[ROWS][72];      // stage derivative
  __shared__ float    sAcc[ROWS][72];    // RK4 accumulator
  __shared__ _Float16 sH1[ROWS][520];    // layer-1 act (row-major, padded)
  __shared__ _Float16 sH2[ROWS][520];    // layer-2 act
  __shared__ float    sTr[ROWS];

  const int tid    = threadIdx.x;
  const int wave   = tid >> 5;
  const int lane   = tid & 31;
  const int halfId = lane >> 4;
  const int lm     = lane & 15;
  const int row0   = blockIdx.x * ROWS;

  for (int idx = tid; idx < ROWS * 65; idx += BLK) {
    int m = idx / 65, c = idx % 65;
    sS[m][c] = (c < CNF_D) ? x[(size_t)(row0 + m) * CNF_D + c] : 0.f;
  }
  __syncthreads();

  for (int step = 0; step < CNF_NSTEPS - 1; ++step) {
    float t0 = tspan[step];
    float dt = tspan[step + 1] - t0;

    for (int idx = tid; idx < ROWS * 65; idx += BLK) {
      int m = idx / 65, c = idx % 65;
      sXi[m][c]  = sS[m][c];
      sAcc[m][c] = 0.f;
    }
    __syncthreads();

    #pragma unroll 1
    for (int stage = 0; stage < 4; ++stage) {
      const float ct   = (stage == 0) ? 0.f : (stage == 3 ? 1.f : 0.5f);
      const float wk   = (stage == 0 || stage == 3) ? 1.f : 2.f;
      const float axm  = (stage < 2) ? 0.5f : 1.f;
      const float tcur = t0 + dt * ct;

      // ---- phase A: h1 = tanh(Xi[:, :64] @ W1x + t*W1[64,:] + b1) ----
      for (int nt = wave; nt < 32; nt += 4) {
        v8f acc = {};
        #pragma unroll
        for (int kt = 0; kt < 2; ++kt) {
          int k0 = kt * 32 + halfId * 8;
          v16h a;
          #pragma unroll
          for (int h = 0; h < 8; ++h) a[h]     = (_Float16)sXi[lm][k0 + h];
          #pragma unroll
          for (int h = 0; h < 8; ++h) a[8 + h] = (_Float16)sXi[lm][k0 + 16 + h];
          const v8h* bp = (const v8h*)(W1B + ((size_t)(kt * 32 + nt) * 512) + lane * 16);
          v16h b = cat16(bp[0], bp[1]);
          acc = __builtin_amdgcn_wmma_f32_16x16x32_f16(false, a, false, b, (short)0, acc, false, false);
        }
        int n = nt * 16 + lm;
        float tb = tcur * W1[CNF_D * CNF_H + n] + b1[n];
        #pragma unroll
        for (int r = 0; r < 8; ++r) {
          int m = r + 8 * halfId;
          float hh = fast_tanh(acc[r] + tb);
          sH1[m][n] = (_Float16)hh;
        }
      }
      __syncthreads();

      // ---- phase B: h2 = tanh(h1 @ W2 + b2) ----
      for (int nt = wave; nt < 32; nt += 4) {
        v8f acc = {};
        #pragma unroll
        for (int kt = 0; kt < 16; ++kt) {
          const v8h* ap = (const v8h*)(&sH1[lm][kt * 32 + halfId * 8]);
          v16h a = cat16(ap[0], ap[2]);
          if (kt + 1 < 16)
            __builtin_prefetch(W2B + ((size_t)((kt + 1) * 32 + nt) * 512) + lane * 16, 0, 3);
          const v8h* bp = (const v8h*)(W2B + ((size_t)(kt * 32 + nt) * 512) + lane * 16);
          v16h b = cat16(bp[0], bp[1]);
          acc = __builtin_amdgcn_wmma_f32_16x16x32_f16(false, a, false, b, (short)0, acc, false, false);
        }
        int n = nt * 16 + lm;
        float b2n = b2[n];
        #pragma unroll
        for (int r = 0; r < 8; ++r) {
          int m = r + 8 * halfId;
          float hh = fast_tanh(acc[r] + b2n);
          sH2[m][n] = (_Float16)hh;
        }
      }
      __syncthreads();

      // ---- phase C: dx = h2 @ W3 + b3 (4 N-tiles, one per wave) ----
      {
        int nt = wave;
        v8f acc = {};
        #pragma unroll
        for (int kt = 0; kt < 16; ++kt) {
          const v8h* ap = (const v8h*)(&sH2[lm][kt * 32 + halfId * 8]);
          v16h a = cat16(ap[0], ap[2]);
          const v8h* bp = (const v8h*)(W3B + ((size_t)(kt * 4 + nt) * 512) + lane * 16);
          v16h b = cat16(bp[0], bp[1]);
          acc = __builtin_amdgcn_wmma_f32_16x16x32_f16(false, a, false, b, (short)0, acc, false, false);
        }
        int n = nt * 16 + lm;
        float b3n = b3[n];
        #pragma unroll
        for (int r = 0; r < 8; ++r) {
          int m = r + 8 * halfId;
          sK[m][n] = acc[r] + b3n;
        }
        if (tid < ROWS) sTr[tid] = 0.f;
      }
      __syncthreads();

      // ---- phase D: tr = (g1 @ C) . g2, g = 1-h^2 recomputed on the fly ----
      {
        const _Float16 one = (_Float16)1.f;
        float trp[8];
        #pragma unroll
        for (int r = 0; r < 8; ++r) trp[r] = 0.f;
        for (int nt = wave; nt < 32; nt += 4) {
          v8f acc = {};
          #pragma unroll
          for (int kt = 0; kt < 16; ++kt) {
            const v8h* ap = (const v8h*)(&sH1[lm][kt * 32 + halfId * 8]);
            v16h hv = cat16(ap[0], ap[2]);
            v16h a;
            #pragma unroll
            for (int i = 0; i < 16; ++i) {     // g1 = 1 - h1^2  (v_pk_fma_f16)
              _Float16 h = hv[i];
              a[i] = one - h * h;
            }
            const v8h* bp = (const v8h*)(CB + ((size_t)(kt * 32 + nt) * 512) + lane * 16);
            v16h b = cat16(bp[0], bp[1]);
            acc = __builtin_amdgcn_wmma_f32_16x16x32_f16(false, a, false, b, (short)0, acc, false, false);
          }
          int n = nt * 16 + lm;
          #pragma unroll
          for (int r = 0; r < 8; ++r) {
            int m = r + 8 * halfId;
            float h2 = (float)sH2[m][n];
            trp[r] += acc[r] * (1.f - h2 * h2);   // g2 on the fly
          }
        }
        #pragma unroll
        for (int r = 0; r < 8; ++r) {
          float v = trp[r];
          v += __shfl_xor(v, 1);
          v += __shfl_xor(v, 2);
          v += __shfl_xor(v, 4);
          v += __shfl_xor(v, 8);
          if (lm == 0) atomicAdd(&sTr[r + 8 * halfId], v);
        }
      }
      __syncthreads();
      if (tid < ROWS) sK[tid][CNF_D] = -sTr[tid];
      __syncthreads();

      // ---- RK4 bookkeeping ----
      for (int idx = tid; idx < ROWS * 65; idx += BLK) {
        int m = idx / 65, c = idx % 65;
        float kv = sK[m][c];
        sAcc[m][c] += wk * kv;
        if (stage < 3) sXi[m][c] = sS[m][c] + axm * dt * kv;
      }
      __syncthreads();
    }

    for (int idx = tid; idx < ROWS * 65; idx += BLK) {
      int m = idx / 65, c = idx % 65;
      sS[m][c] += (dt / 6.f) * sAcc[m][c];
    }
    __syncthreads();
  }

  for (int idx = tid; idx < ROWS * CNF_D; idx += BLK) {
    int m = idx / CNF_D, c = idx % CNF_D;
    outZ[(size_t)(row0 + m) * CNF_D + c] = sS[m][c];
  }
  if (tid < ROWS) outL[row0 + tid] = sS[tid][CNF_D];
}

extern "C" void kernel_launch(void* const* d_in, const int* in_sizes, int n_in,
                              void* d_out, int out_size, void* d_ws, size_t ws_size,
                              hipStream_t stream) {
  (void)in_sizes; (void)n_in; (void)out_size; (void)ws_size;
  const float* x  = (const float*)d_in[0];
  const float* ts = (const float*)d_in[1];
  const float* W1 = (const float*)d_in[2];
  const float* b1 = (const float*)d_in[3];
  const float* W2 = (const float*)d_in[4];
  const float* b2 = (const float*)d_in[5];
  const float* W3 = (const float*)d_in[6];
  const float* b3 = (const float*)d_in[7];
  float* outZ = (float*)d_out;
  float* outL = outZ + (size_t)CNF_B * CNF_D;

  // workspace: f16 swizzled weights (~1.13 MB total)
  _Float16* W1B = (_Float16*)d_ws;          // 64*512
  _Float16* W2B = W1B + 64 * 512;           // 512*512
  _Float16* W3B = W2B + 512 * 512;          // 512*64
  _Float16* CB  = W3B + 512 * 64;           // 512*512

  cnf_prep_swizzle<<<(64 * 512 + 255) / 256, 256, 0, stream>>>(W1, W1B, 64, 512);
  cnf_prep_swizzle<<<(512 * 512 + 255) / 256, 256, 0, stream>>>(W2, W2B, 512, 512);
  cnf_prep_swizzle<<<(512 * 64 + 255) / 256, 256, 0, stream>>>(W3, W3B, 512, 64);
  cnf_prep_cmat<<<(512 * 512 + 255) / 256, 256, 0, stream>>>(W1, W2, W3, CB);

  cnf_ode_kernel<<<CNF_B / ROWS, BLK, 0, stream>>>(x, ts, W1, b1, b2, b3,
                                                   W1B, W2B, W3B, CB, outZ, outL);
}
